// RETAIN_39539468927539
// MI455X (gfx1250) — matrix operand order repaired and checked by reading, not compiled, not run on hardware
//
#include <hip/hip_runtime.h>

// ---------------------------------------------------------------------------
// Problem constants (from reference): B=64, S=512, VOCAB=1400, HID=256, NCLS=128
// ---------------------------------------------------------------------------
#define B_    64
#define S_    512
#define V_    1400
#define H_    256
#define G_    1024          // 4*H
#define NCLS_ 128
#define BS_   (B_*S_)       // 32768

typedef unsigned short ushort_t;
typedef __attribute__((ext_vector_type(16))) __bf16 v16bf;
typedef __attribute__((ext_vector_type(8)))  float  v8f;
typedef __attribute__((ext_vector_type(4)))  unsigned v4u;

union Frag { v16bf v; unsigned u[8]; v4u q[2]; };

__device__ __forceinline__ float bf2f(ushort_t h){
  union { unsigned u; float f; } x; x.u = ((unsigned)h) << 16; return x.f;
}
__device__ __forceinline__ ushort_t f2bf(float f){
  union { float f; unsigned u; } x; x.f = f;
  unsigned u = x.u;
  return (ushort_t)((u + 0x7FFFu + ((u >> 16) & 1u)) >> 16);   // RNE
}
__device__ __forceinline__ unsigned packbf(float a, float b){
  return (unsigned)f2bf(a) | ((unsigned)f2bf(b) << 16);
}
__device__ __forceinline__ float fsig(float x){ return 1.f / (1.f + __expf(-x)); }
__device__ __forceinline__ float ftanh_(float x){
  x = fminf(fmaxf(x, -15.f), 15.f);
  float e = __expf(2.f * x);
  return (e - 1.f) / (e + 1.f);
}
__device__ __forceinline__ v8f vzero(){ v8f z; for (int i = 0; i < 8; i++) z[i] = 0.f; return z; }

__device__ __forceinline__ v8f wmma_bf(const Frag& a, const Frag& b, v8f c){
  // (neg_a, A, neg_b, B, c_mod, C, reuse_a, reuse_b)
  return __builtin_amdgcn_wmma_f32_16x16x32_bf16(false, a.v, false, b.v, (short)0, c, false, false);
}
__device__ __forceinline__ void load_frag2x16(Frag& f, const unsigned* p0, const unsigned* p1){
  f.q[0] = *(const v4u*)p0;     // 16B
  f.q[1] = *(const v4u*)p1;     // 16B
}

// ---------------------------------------------------------------------------
// Pack a B operand into FRAGMENT-MAJOR bf16x2 layout:
//   dst[ ((kp>>3)*N + n)*8 + (kp&7) ]  packs k=2kp (lo16), k=2kp+1 (hi16).
// A WMMA B-fragment (8 pair-rows at one column) is then 32 contiguous bytes
// -> loads as two b128s. transpose=1: logical B[k][n] = src[n*ldsrc + k].
// ---------------------------------------------------------------------------
__global__ void pack_b_kernel(const float* __restrict__ src, unsigned* __restrict__ dst,
                              int K, int N, int ldsrc, int transpose){
  int idx = blockIdx.x * blockDim.x + threadIdx.x;
  int Kp = (K + 1) >> 1;
  if (idx >= Kp * N) return;
  int kp = idx / N, n = idx - kp * N;
  int k0 = 2 * kp;
  float a = transpose ? src[(size_t)n * ldsrc + k0] : src[(size_t)k0 * ldsrc + n];
  float b = 0.f;
  if (k0 + 1 < K)
    b = transpose ? src[(size_t)n * ldsrc + k0 + 1] : src[(size_t)(k0 + 1) * ldsrc + n];
  dst[(((size_t)(kp >> 3)) * N + n) * 8 + (kp & 7)] = packbf(a, b);
}

// ---------------------------------------------------------------------------
// Generic bf16 WMMA GEMM: C[M,N] = act(A[M,K] @ B[K,N]).
// A: fp32 (converted during LDS staging) or bf16 row-major. B: fragment-major.
// Block tile 128x64, 8 waves of 32x32, K-chunk 32.
// ---------------------------------------------------------------------------
template<bool A_F32, bool TANH_ACT, bool OUT_BF16>
__global__ __launch_bounds__(256) void gemm_wmma_kernel(
    const void* __restrict__ Av, const unsigned* __restrict__ BF, void* __restrict__ Cv,
    int M, int N, int K, int lda, int ldc)
{
  __shared__ unsigned As[128 * 16];   // [row][kpair] bf16x2 : 8 KB
  __shared__ unsigned Bs[16 * 64];    // fragment-major [g][col][j] : 4 KB

  const int tid = threadIdx.x, lane = tid & 31, w = tid >> 5;
  const int mwave = w & 3, nwave = w >> 2;
  const int m0 = blockIdx.x * 128, n0 = blockIdx.y * 64;
  const int row16 = lane & 15, half = lane >> 4;
  const int Kp = (K + 1) >> 1;
  const int nk = (K + 31) >> 5;
  const float*    Af = (const float*)Av;
  const ushort_t* Ab = (const ushort_t*)Av;

  v8f acc[2][2];
  for (int mt = 0; mt < 2; mt++) for (int nt = 0; nt < 2; nt++) acc[mt][nt] = vzero();

  #pragma unroll 1
  for (int kc = 0; kc < nk; kc++){
    // ---- stage A (convert fp32 -> bf16 pairs if needed) ----
    for (int i = tid; i < 128 * 16; i += 256){
      int r = i >> 4, kp = i & 15;
      int grow = m0 + r, k0 = kc * 32 + 2 * kp;
      unsigned val = 0u;
      if (grow < M){
        if (A_F32){
          float a0 = (k0     < K) ? Af[(size_t)grow * lda + k0]     : 0.f;
          float a1 = (k0 + 1 < K) ? Af[(size_t)grow * lda + k0 + 1] : 0.f;
          val = packbf(a0, a1);
        } else {
          if      (k0 + 1 < K) val = *(const unsigned*)(Ab + (size_t)grow * lda + k0);
          else if (k0     < K) val = (unsigned)Ab[(size_t)grow * lda + k0];
        }
      }
      As[i] = val;
    }
    // ---- stage B (fragment-major in both global and LDS) ----
    for (int i = tid; i < 16 * 64; i += 256){
      int j = i & 7, c = (i >> 3) & 63, g = i >> 9;     // i = (g*64+c)*8+j
      int kp = kc * 16 + g * 8 + j, col = n0 + c;
      Bs[i] = (kp < Kp && col < N) ? BF[(((size_t)(kc * 2 + g)) * N + col) * 8 + j] : 0u;
    }
    __syncthreads();
    if (kc + 1 < nk)
      __builtin_prefetch(BF + (((size_t)(kc * 2 + 2)) * N + n0) * 8 + lane, 0, 1);

    Frag af[2], bf[2];
    for (int mt = 0; mt < 2; mt++){
      int r = mwave * 32 + mt * 16 + row16;
      const unsigned* p = &As[r * 16 + half * 8];   // K split by lane half: k = 16*half + s (contiguous), same as B
      load_frag2x16(af[mt], p, p + 4);
    }
    for (int nt = 0; nt < 2; nt++){
      int c = nwave * 32 + nt * 16 + row16;
      const unsigned* p = &Bs[(half * 64 + c) * 8];
      load_frag2x16(bf[nt], p, p + 4);
    }
    for (int mt = 0; mt < 2; mt++)
      for (int nt = 0; nt < 2; nt++)
        acc[mt][nt] = wmma_bf(af[mt], bf[nt], acc[mt][nt]);
    __syncthreads();
  }

  // ---- epilogue: C layout VGPR e -> M=e (+8 for upper lane half), lane%16 -> N ----
  for (int mt = 0; mt < 2; mt++)
    for (int nt = 0; nt < 2; nt++)
      for (int e = 0; e < 8; e++){
        int r = m0 + mwave * 32 + mt * 16 + e + 8 * half;
        int c = n0 + nwave * 32 + nt * 16 + row16;
        if (r < M && c < N){
          float v = acc[mt][nt][e];
          if (TANH_ACT) v = ftanh_(v);
          if (OUT_BF16) ((ushort_t*)Cv)[(size_t)r * ldc + c] = f2bf(v);
          else          ((float*)   Cv)[(size_t)r * ldc + c] = v;
        }
      }
}

// ---------------------------------------------------------------------------
// Persistent time-LSTM: grid = 2 blocks (one per LSTM), 1024 threads = 32 waves.
// Each wave owns 2 (m-tile, j-tile) tasks -> ~150 live VGPRs, no spills.
// h,c bf16 in LDS (A operands); fp32 c master copy lives in registers across
// all 512 steps. Gate tiles f/i/o/c~ for column j sit at n = j,H+j,2H+j,3H+j,
// so the nonlinear combine is register-only. Fragment-major W.T streams from
// L2 every step (512 KB, L2-resident) as two b128 loads per fragment.
// kc loops kept rolled so the compiler doesn't preload+spill B fragments.
// ---------------------------------------------------------------------------
__global__ __launch_bounds__(1024) void tlstm_kernel(
    const unsigned* __restrict__ W1F, const unsigned* __restrict__ Wd1F,
    const ushort_t* __restrict__ xU1,
    const float* __restrict__ ball1, const float* __restrict__ bu1, const float* __restrict__ bd1,
    ushort_t* __restrict__ out1,
    const unsigned* __restrict__ W2F, const unsigned* __restrict__ Wd2F,
    const ushort_t* __restrict__ xU2,
    const float* __restrict__ ball2, const float* __restrict__ bu2, const float* __restrict__ bd2,
    ushort_t* __restrict__ out2,
    const float* __restrict__ ts)
{
  const int L = blockIdx.x;
  const unsigned* WF   = L ? W2F   : W1F;
  const unsigned* WdF  = L ? Wd2F  : Wd1F;
  const ushort_t* xU   = L ? xU2   : xU1;
  const float*    ball = L ? ball2 : ball1;
  const float*    bu   = L ? bu2   : bu1;
  const float*    bd   = L ? bd2   : bd1;
  ushort_t*       out  = L ? out2  : out1;

  __shared__ ushort_t hS[64 * 256];   // 32 KB
  __shared__ ushort_t cS[64 * 256];   // 32 KB
  __shared__ float    tS[64];

  const int tid = threadIdx.x, lane = tid & 31, w = tid >> 5;
  const int m = w >> 3, jq = w & 7;    // wave -> (m-tile, group of 2 j-tiles)
  const int row16 = lane & 15, half = lane >> 4;

  for (int i = tid; i < 64 * 256; i += 1024){ hS[i] = 0; cS[i] = 0; }
  v8f creg[2];
  for (int q = 0; q < 2; q++) creg[q] = vzero();
  __syncthreads();

  const unsigned* hU = (const unsigned*)hS;
  const unsigned* cU = (const unsigned*)cS;
  // A-fragment base (uint index): r*128 + kc*16 + half*8  (16B aligned)
  const int abase = (m * 16 + row16) * 128 + half * 8;

  for (int t = 0; t < S_; t++){
    if (tid < 64) tS[tid] = ts[(size_t)tid * S_ + t];
    __syncthreads();

    // ---- phase 1: c_s1 = tanh(c @ Wd.T + bd); c_adj = c + c_s1*(t-1) ----
    v8f cadj[2];
    {
      v8f acc[2];
      for (int q = 0; q < 2; q++) acc[q] = vzero();
      #pragma unroll 1
      for (int kc = 0; kc < 8; kc++){
        Frag a;
        { const unsigned* p = cU + abase + kc * 16;
          load_frag2x16(a, p, p + 4); }
        for (int q = 0; q < 2; q++){
          int c = (jq * 2 + q) * 16 + row16;
          Frag b;
          const unsigned* p = WdF + (((size_t)(kc * 2 + half)) * H_ + c) * 8;
          load_frag2x16(b, p, p + 4);
          acc[q] = wmma_bf(a, b, acc[q]);
        }
      }
      for (int q = 0; q < 2; q++){
        int c = (jq * 2 + q) * 16 + row16;
        float bdv = bd[c];
        for (int e = 0; e < 8; e++){
          int rb = m * 16 + e + 8 * half;
          float cs1 = ftanh_(acc[q][e] + bdv);
          cadj[q][e] = creg[q][e] + cs1 * (tS[rb] - 1.f);
        }
      }
    }

    // ---- phase 2: gates = h @ W_all.T  (4 gate tiles per (m,j) task) ----
    v8f gacc[2][4];
    for (int q = 0; q < 2; q++) for (int g = 0; g < 4; g++) gacc[q][g] = vzero();
    #pragma unroll 1
    for (int kc = 0; kc < 8; kc++){
      Frag a;
      { const unsigned* p = hU + abase + kc * 16;
        load_frag2x16(a, p, p + 4); }
      for (int q = 0; q < 2; q++){
        int j0 = (jq * 2 + q) * 16;
        for (int g = 0; g < 4; g++){
          int cg = g * H_ + j0 + row16;
          Frag b;
          const unsigned* p = WF + (((size_t)(kc * 2 + half)) * G_ + cg) * 8;
          load_frag2x16(b, p, p + 4);
          gacc[q][g] = wmma_bf(a, b, gacc[q][g]);
        }
      }
    }
    __syncthreads();   // all waves done reading hS/cS before overwrite

    // ---- epilogue: sigmoid gates + xU + biases, cell/hidden update ----
    for (int q = 0; q < 2; q++){
      int c = (jq * 2 + q) * 16 + row16;
      float bsf = ball[c]          + bu[c];
      float bsi = ball[H_ + c]     + bu[H_ + c];
      float bso = ball[2 * H_ + c] + bu[2 * H_ + c];
      float bsc = ball[3 * H_ + c] + bu[3 * H_ + c];
      for (int e = 0; e < 8; e++){
        int rb = m * 16 + e + 8 * half;
        size_t xb = ((size_t)rb * S_ + t) * G_;
        float fg = fsig(gacc[q][0][e] + bsf + bf2f(xU[xb + c]));
        float ig = fsig(gacc[q][1][e] + bsi + bf2f(xU[xb + H_ + c]));
        float og = fsig(gacc[q][2][e] + bso + bf2f(xU[xb + 2 * H_ + c]));
        float cg = fsig(gacc[q][3][e] + bsc + bf2f(xU[xb + 3 * H_ + c]));
        float cn = fg * cadj[q][e] + ig * cg;
        creg[q][e] = cn;
        float hn = og * ftanh_(cn);
        ushort_t hb = f2bf(hn);
        hS[rb * 256 + c] = hb;
        cS[rb * 256 + c] = f2bf(cn);
        out[((size_t)rb * S_ + t) * H_ + c] = hb;
      }
    }
    __syncthreads();
  }
}

// ---------------------------------------------------------------------------
// alpha[b,s] = softmax_s( out1[b,s,:] . wa )
// ---------------------------------------------------------------------------
__global__ __launch_bounds__(256) void alpha_kernel(const ushort_t* __restrict__ out1,
                                                    const float* __restrict__ wa,
                                                    float* __restrict__ alpha){
  __shared__ float sc[512];
  __shared__ float red[256];
  int b = blockIdx.x, t = threadIdx.x;
  for (int s = t; s < 512; s += 256){
    const ushort_t* r = out1 + ((size_t)b * 512 + s) * 256;
    float acc = 0.f;
    for (int h = 0; h < 256; h++) acc += bf2f(r[h]) * wa[h];
    sc[s] = acc;
  }
  __syncthreads();
  red[t] = fmaxf(sc[t], sc[t + 256]);
  __syncthreads();
  for (int off = 128; off > 0; off >>= 1){ if (t < off) red[t] = fmaxf(red[t], red[t + off]); __syncthreads(); }
  float mx = red[0];
  __syncthreads();
  float e0 = __expf(sc[t] - mx), e1 = __expf(sc[t + 256] - mx);
  red[t] = e0 + e1;
  __syncthreads();
  for (int off = 128; off > 0; off >>= 1){ if (t < off) red[t] += red[t + off]; __syncthreads(); }
  float inv = 1.f / red[0];
  alpha[(size_t)b * 512 + t]       = e0 * inv;
  alpha[(size_t)b * 512 + t + 256] = e1 * inv;
}

// ---------------------------------------------------------------------------
// ctx[b,h] = sum_s emb[b,s,h] * Beta[b,s,h] * alpha[b,s]
// ---------------------------------------------------------------------------
__global__ __launch_bounds__(256) void ctx_reduce_kernel(const ushort_t* __restrict__ embB,
                                                         const float* __restrict__ Beta,
                                                         const float* __restrict__ alpha,
                                                         float* __restrict__ ctx){
  int b = blockIdx.x, h = threadIdx.x;
  float acc = 0.f;
  for (int s = 0; s < 512; s++){
    size_t base = ((size_t)b * 512 + s) * 256 + h;
    acc += bf2f(embB[base]) * Beta[base] * alpha[(size_t)b * 512 + s];
  }
  ctx[(size_t)b * 256 + h] = acc;
}

// ---------------------------------------------------------------------------
extern "C" void kernel_launch(void* const* d_in, const int* in_sizes, int n_in,
                              void* d_out, int out_size, void* d_ws, size_t ws_size,
                              hipStream_t stream) {
  const float* inputs = (const float*)d_in[0];
  const float* ts     = (const float*)d_in[1];
  const float* emb    = (const float*)d_in[2];
  const float* W_all1 = (const float*)d_in[3];  const float* b_all1 = (const float*)d_in[4];
  const float* U_all1 = (const float*)d_in[5];  const float* b_u1   = (const float*)d_in[6];
  const float* W_d1   = (const float*)d_in[7];  const float* b_d1   = (const float*)d_in[8];
  const float* W_all2 = (const float*)d_in[9];  const float* b_all2 = (const float*)d_in[10];
  const float* U_all2 = (const float*)d_in[11]; const float* b_u2   = (const float*)d_in[12];
  const float* W_d2   = (const float*)d_in[13]; const float* b_d2   = (const float*)d_in[14];
  const float* wa     = (const float*)d_in[15];
  const float* Wb     = (const float*)d_in[16];
  const float* W_out  = (const float*)d_in[17];
  float* outp = (float*)d_out;
  char* ws = (char*)d_ws;

  auto alignup = [](size_t x){ return (x + 255) & ~(size_t)255; };
  size_t off = 0;
  auto take = [&](size_t bytes)->char*{ char* p = ws + off; off = alignup(off + bytes); return p; };
  // fragment-major packed size: ceil(Kp/8) groups of 8 pair-rows
  auto packed_bytes = [](int K, int N)->size_t{
    int Kp = (K + 1) >> 1;
    size_t groups = (size_t)((Kp + 7) >> 3);
    return groups * 8 * (size_t)N * 4;
  };

  ushort_t* embB = (ushort_t*)take((size_t)BS_ * H_ * 2);   // embedded, bf16 [BS,H]
  ushort_t* xU1  = (ushort_t*)take((size_t)BS_ * G_ * 2);   // x @ U1.T, bf16 [BS,G]
  ushort_t* xU2  = (ushort_t*)take((size_t)BS_ * G_ * 2);
  ushort_t* o1   = (ushort_t*)take((size_t)BS_ * H_ * 2);   // LSTM1 hidden seq
  ushort_t* o2   = (ushort_t*)take((size_t)BS_ * H_ * 2);
  float*    Beta = (float*)   take((size_t)BS_ * H_ * 4);
  float*    alph = (float*)   take((size_t)B_ * S_ * 4);
  float*    ctxb = (float*)   take((size_t)B_ * H_ * 4);
  unsigned* embP = (unsigned*)take(packed_bytes(V_, H_));
  unsigned* U1P  = (unsigned*)take(packed_bytes(H_, G_));
  unsigned* W1P  = (unsigned*)take(packed_bytes(H_, G_));
  unsigned* Wd1P = (unsigned*)take(packed_bytes(H_, H_));
  unsigned* U2P  = (unsigned*)take(packed_bytes(H_, G_));
  unsigned* W2P  = (unsigned*)take(packed_bytes(H_, G_));
  unsigned* Wd2P = (unsigned*)take(packed_bytes(H_, H_));
  unsigned* WbP  = (unsigned*)take(packed_bytes(H_, H_));
  unsigned* WoP  = (unsigned*)take(packed_bytes(H_, NCLS_));
  (void)ws_size; (void)in_sizes; (void)n_in; (void)out_size;

  // ---- pack all B operands (bf16, fragment-major) ----
  auto g1d = [](int n){ return dim3((unsigned)((n + 255) / 256)); };
  pack_b_kernel<<<g1d((V_/2)*H_),      256, 0, stream>>>(emb,    embP, V_, H_,    H_, 0);
  pack_b_kernel<<<g1d((H_/2)*G_),      256, 0, stream>>>(U_all1, U1P,  H_, G_,    H_, 1);
  pack_b_kernel<<<g1d((H_/2)*G_),      256, 0, stream>>>(W_all1, W1P,  H_, G_,    H_, 1);
  pack_b_kernel<<<g1d((H_/2)*H_),      256, 0, stream>>>(W_d1,   Wd1P, H_, H_,    H_, 1);
  pack_b_kernel<<<g1d((H_/2)*G_),      256, 0, stream>>>(U_all2, U2P,  H_, G_,    H_, 1);
  pack_b_kernel<<<g1d((H_/2)*G_),      256, 0, stream>>>(W_all2, W2P,  H_, G_,    H_, 1);
  pack_b_kernel<<<g1d((H_/2)*H_),      256, 0, stream>>>(W_d2,   Wd2P, H_, H_,    H_, 1);
  pack_b_kernel<<<g1d((H_/2)*H_),      256, 0, stream>>>(Wb,     WbP,  H_, H_,    H_, 1);
  pack_b_kernel<<<g1d((H_/2)*NCLS_),   256, 0, stream>>>(W_out,  WoP,  H_, NCLS_, H_, 1);

  dim3 blk(256);
  // embedded = inputs @ emb  (A fp32, out bf16)
  gemm_wmma_kernel<true, false, true><<<dim3((BS_+127)/128, (H_+63)/64), blk, 0, stream>>>(
      inputs, embP, embB, BS_, H_, V_, V_, H_);
  // xU = embedded @ U_all.T  (A bf16, out bf16), hoisted out of the scan
  gemm_wmma_kernel<false, false, true><<<dim3((BS_+127)/128, (G_+63)/64), blk, 0, stream>>>(
      embB, U1P, xU1, BS_, G_, H_, H_, G_);
  gemm_wmma_kernel<false, false, true><<<dim3((BS_+127)/128, (G_+63)/64), blk, 0, stream>>>(
      embB, U2P, xU2, BS_, G_, H_, H_, G_);
  // recurrent scan: 2 persistent workgroups (one per LSTM), 32 waves each
  tlstm_kernel<<<2, 1024, 0, stream>>>(W1P, Wd1P, xU1, b_all1, b_u1, b_d1, o1,
                                       W2P, Wd2P, xU2, b_all2, b_u2, b_d2, o2, ts);
  // attention
  alpha_kernel<<<B_, 256, 0, stream>>>(o1, wa, alph);
  gemm_wmma_kernel<false, true, false><<<dim3((BS_+127)/128, (H_+63)/64), blk, 0, stream>>>(
      o2, WbP, Beta, BS_, H_, H_, H_, H_);
  ctx_reduce_kernel<<<B_, 256, 0, stream>>>(embB, Beta, alph, ctxb);
  // final projection
  gemm_wmma_kernel<true, false, false><<<dim3(1, (NCLS_+63)/64), blk, 0, stream>>>(
      ctxb, WoP, outp, B_, NCLS_, H_, H_, NCLS_);
}
